// PG_Tree_74852690034877
// MI455X (gfx1250) — matrix operand order repaired and checked by reading, not compiled, not run on hardware
//
#include <hip/hip_runtime.h>

// ---------------------------------------------------------------------------
// PG_Tree forward for MI455X (gfx1250, wave32).
// Bandwidth-bound pipeline; MLP layer-1 uses v_wmma_f32_16x16x32_bf16 with
// bf16 activations (f32 accumulate), W1 pre-swizzled into WMMA B-fragment
// layout and staged in LDS per block via TDM / async-LDS when available.
// ---------------------------------------------------------------------------

#define NPTS   131072
#define FDIM   63
#define FPAD   64          // padded feature dim (f32 buffers, d_out rows)
#define F2PAD  128         // padded MLP input dim (bf16 feats)
#define HID    32
#define NLEV   13          // MAXD + 1
#define MAXN   32
#define KMAX   8192        // max live subtrees (true bound: N/MAXN = 4096)
#define CMAX   16384       // max children per level
#define EMAX   (1 << 24)   // edge buffer capacity
#define BN_EPS 1e-5f

// ctrl integer slots
#define CI_NODE    0
#define CI_EDGE    1
#define CI_NUMSEG  2
#define CI_NUMLEAF 3
#define CI_K2      4
#define CI_LVLBASE 5

// -------- CDNA5 data-mover feature detection (compile-safe guards) ---------
#if defined(__HIP_DEVICE_COMPILE__) && defined(__has_builtin)
#  if __has_builtin(__builtin_amdgcn_tensor_load_to_lds) && \
      __has_builtin(__builtin_amdgcn_s_wait_tensorcnt)
#    define USE_TDM 1
#  endif
#  if !defined(USE_TDM) && \
      __has_builtin(__builtin_amdgcn_global_load_async_to_lds_b128) && \
      __has_builtin(__builtin_amdgcn_s_wait_asynccnt)
#    define USE_ASYNC 1
#  endif
#endif
#ifndef USE_TDM
#  define USE_TDM 0
#endif
#ifndef USE_ASYNC
#  define USE_ASYNC 0
#endif

typedef __bf16 v16bf __attribute__((ext_vector_type(16)));
typedef float  v8f   __attribute__((ext_vector_type(8)));
typedef unsigned int u32x4 __attribute__((ext_vector_type(4)));
typedef int          i32x4 __attribute__((ext_vector_type(4)));
typedef int          i32x8 __attribute__((ext_vector_type(8)));

union BFrag { uint4 q[2]; v16bf v; };

__device__ __forceinline__ unsigned short f2bf(float x) {
  union { float f; unsigned u; } c; c.f = x;
  unsigned r = c.u + 0x7FFFu + ((c.u >> 16) & 1u);   // round-to-nearest-even
  return (unsigned short)(r >> 16);
}
__device__ __forceinline__ float seluf(float x) {
  const float sc = 1.0507009873554805f, al = 1.6732632423543772f;
  return x > 0.f ? sc * x : sc * al * (__expf(x) - 1.f);
}
__device__ __forceinline__ float sigmoidf_(float x) { return 1.f / (1.f + __expf(-x)); }

// ---------------------------------------------------------------------------
// BatchNorm statistics: one block per feature, strided row reduction.
// ---------------------------------------------------------------------------
__global__ void bn_reduce(const float* __restrict__ X,
                          float* __restrict__ bnSum, float* __restrict__ bnSq) {
  int f = blockIdx.x;
  if (f >= FDIM) return;
  __shared__ float s1[256], s2[256];
  float a = 0.f, b = 0.f;
  for (int i = threadIdx.x; i < NPTS; i += 256) {
    float x = X[(size_t)i * FDIM + f];
    a += x; b += x * x;
  }
  s1[threadIdx.x] = a; s2[threadIdx.x] = b;
  __syncthreads();
  for (int st = 128; st > 0; st >>= 1) {
    if (threadIdx.x < st) {
      s1[threadIdx.x] += s1[threadIdx.x + st];
      s2[threadIdx.x] += s2[threadIdx.x + st];
    }
    __syncthreads();
  }
  if (threadIdx.x == 0) { bnSum[f] = s1[0]; bnSq[f] = s2[0]; }
}

// ---------------------------------------------------------------------------
// Normalize; write f32 X_norm, bf16 feats cols [0,63], and d_out X block
// (original points get appended zero column, col 63).
// ---------------------------------------------------------------------------
__global__ void bn_normalize(const float* __restrict__ X,
                             const float* __restrict__ bnSum, const float* __restrict__ bnSq,
                             const float* __restrict__ g, const float* __restrict__ bta,
                             float* __restrict__ Xn, unsigned short* __restrict__ feats,
                             float* __restrict__ dout) {
  size_t t = (size_t)blockIdx.x * blockDim.x + threadIdx.x;
  if (t >= (size_t)NPTS * FPAD) return;
  int f = (int)(t & (FPAD - 1));
  size_t i = t >> 6;
  float v = 0.f;
  if (f < FDIM) {
    float m   = bnSum[f] * (1.f / NPTS);
    float var = bnSq[f] * (1.f / NPTS) - m * m;
    float x   = X[i * FDIM + f];
    v = (x - m) * rsqrtf(var + BN_EPS) * g[f] + bta[f];
  }
  Xn[i * FPAD + f]    = v;
  feats[i * F2PAD + f] = f2bf(v);
  dout[i * FPAD + f]  = (f < FDIM) ? v : 0.f;
}

__global__ void init_state(int* __restrict__ seg, float* __restrict__ mask,
                           int* __restrict__ ctrlI, float* __restrict__ ctrlF) {
  size_t t = (size_t)blockIdx.x * blockDim.x + threadIdx.x;
  if (t < NPTS) { seg[t] = 0; mask[t] = 1.f; }
  if (t == 0) {
    ctrlI[CI_NODE] = NPTS; ctrlI[CI_EDGE] = 0; ctrlI[CI_NUMSEG] = 1;
    ctrlI[CI_NUMLEAF] = 0; ctrlI[CI_K2] = 0; ctrlI[CI_LVLBASE] = 0;
    ctrlF[0] = 0.f;
  }
}

// ---------------------------------------------------------------------------
// Pre-swizzle W1 (126x32 f32, K-padded to 128) into the bf16 B-fragment
// layout of V_WMMA_F32_16X16X32_BF16: fragment (k, ntile) holds B[k*32 ..][nt*16..],
// lane l: N = (l&15)+nt*16, element j -> K = k*32 + (l>>4)*8 + (j<8 ? j : 16+(j-8)).
// ---------------------------------------------------------------------------
__global__ void prep_w1frag(const float* __restrict__ W1, unsigned short* __restrict__ frag) {
  int t = blockIdx.x * blockDim.x + threadIdx.x;
  if (t >= 4 * 2 * 32 * 16) return;
  int j    = t & 15;
  int lane = (t >> 4) & 31;
  int nt   = (t >> 9) & 1;
  int k    = t >> 10;
  int sel  = lane >> 4;
  int n    = nt * 16 + (lane & 15);
  int K    = k * 32 + sel * 8 + (j < 8 ? j : 16 + (j - 8));
  float v  = (K < 2 * FDIM) ? W1[(size_t)K * HID + n] : 0.f;
  frag[t] = f2bf(v);
}

// ---------------------------------------------------------------------------
// Per-level segment sums (means of current subtrees).
// ---------------------------------------------------------------------------
__global__ void seg_accum(const float* __restrict__ Xn, const int* __restrict__ seg,
                          float* __restrict__ segSum, int* __restrict__ segCnt) {
  int i = blockIdx.x * 256 + threadIdx.x;
  if (i >= NPTS) return;
  int s = seg[i];
  if (s < 0) return;
  atomicAdd(&segCnt[s], 1);
  const float* x = &Xn[(size_t)i * FPAD];
  float* o = &segSum[(size_t)s * FPAD];
  for (int f = 0; f < FDIM; ++f) atomicAdd(&o[f], x[f]);
}

__global__ void level_node_base(int* __restrict__ ctrlI) {
  if (threadIdx.x == 0 && blockIdx.x == 0) {
    ctrlI[CI_LVLBASE] = ctrlI[CI_NODE];
    ctrlI[CI_NODE]   += ctrlI[CI_NUMSEG];
  }
}

// one new node per live subtree: its mean, with appended one column
__global__ void emit_level_nodes(const float* __restrict__ segSum, const int* __restrict__ segCnt,
                                 const int* __restrict__ ctrlI, int* __restrict__ levelNode,
                                 float* __restrict__ dout) {
  size_t t = (size_t)blockIdx.x * blockDim.x + threadIdx.x;
  int f = (int)(t & 63);
  int k = (int)(t >> 6);
  if (k >= ctrlI[CI_NUMSEG]) return;
  int row = ctrlI[CI_LVLBASE] + k;
  if (f == 0) levelNode[k] = row;
  int   c = segCnt[k] > 0 ? segCnt[k] : 1;
  float v = (f < FDIM) ? segSum[(size_t)k * FPAD + f] / (float)c : 1.f;
  dout[(size_t)row * FPAD + f] = v;
}

// agg columns of the MLP input (zeros for inactive points, as in reference)
__global__ void agg_write(const float* __restrict__ segSum, const int* __restrict__ segCnt,
                          const int* __restrict__ seg, unsigned short* __restrict__ feats) {
  size_t t = (size_t)blockIdx.x * blockDim.x + threadIdx.x;
  if (t >= (size_t)NPTS * FPAD) return;
  int f = (int)(t & 63);
  size_t i = t >> 6;
  int s = seg[i];
  float v = 0.f;
  if (s >= 0 && f < FDIM) {
    int c = segCnt[s] > 0 ? segCnt[s] : 1;
    v = segSum[(size_t)s * FPAD + f] / (float)c;
  }
  feats[i * F2PAD + FPAD + f] = f2bf(v);
}

// ---------------------------------------------------------------------------
// Fused MLP: layer1 (128->32) via WMMA bf16, SELU, layer2 (32->1) via VALU dot
// + 16-lane butterfly reduction, sigmoid, split decision, masked pg accumulate.
// One wave owns a 16-row output tile; block = 8 waves = 128 rows.
// W1 fragments (8 KB) staged to LDS via TDM (tensor_load_to_lds) or async-LDS
// loads when the toolchain exposes them; manual fallback otherwise.
// ---------------------------------------------------------------------------
__global__ __launch_bounds__(256) void mlp_wmma(
    const unsigned short* __restrict__ feats, const unsigned short* __restrict__ w1frag,
    const float* __restrict__ b1, const float* __restrict__ W2, const float* __restrict__ b2,
    const float* __restrict__ ud, const float* __restrict__ mask,
    float* __restrict__ probs, int* __restrict__ splits, float* __restrict__ pgAcc) {
  __shared__ __align__(16) unsigned short w1s[4096];   // 8 KB of B-fragments
  int tid = threadIdx.x;

#if USE_TDM
  // ---- Tensor Data Mover: one D# describing an 8 KB contiguous tile -------
  // data_size=8B (code 3); tensor_dim0 = tile_dim0 = 1024 units; 1 row.
  if (tid < 32) {
    unsigned long long ga = (unsigned long long)(const void*)w1frag;
    unsigned lds = (unsigned)(unsigned long long)(void*)w1s;   // LDS address: low 32 bits of pointer
    u32x4 g0;
    g0.x = 1u;                                      // count=1, user descriptor
    g0.y = lds;                                     // lds_addr  (bits 63:32)
    g0.z = (unsigned)(ga & 0xFFFFFFFFu);            // global_addr[31:0]
    g0.w = (unsigned)((ga >> 32) & 0x01FFFFFFu)     // global_addr[56:32]
         | 0x80000000u;                             // type=2 ("image", bits 127:126)
    i32x8 g1;
    g1[0] = (int)(3u << 16);                        // data_size=8B, no multicast
    g1[1] = (int)(1024u << 16);                     // tensor_dim0[15:0] @ bits 63:48
    g1[2] = (int)(1u << 16);                        // tensor_dim0 hi = 0, tensor_dim1 = 1
    g1[3] = (int)(1024u << 16);                     // tensor_dim1 hi = 0, tile_dim0 = 1024
    g1[4] = 1;                                      // tile_dim1 = 1, tile_dim2 = 0
    g1[5] = 1024;                                   // tensor_dim0_stride low32
    g1[6] = (int)(1024u << 16);                     // stride0 hi16=0, stride1 lo16
    g1[7] = 0;                                      // stride1 hi32
    i32x4 gz = {0, 0, 0, 0};
#if __clang_major__ >= 23
    i32x8 gz8 = {0, 0, 0, 0, 0, 0, 0, 0};
    __builtin_amdgcn_tensor_load_to_lds(g0, g1, gz, gz, gz8, 0);
#else
    __builtin_amdgcn_tensor_load_to_lds(g0, g1, gz, gz, 0);
#endif
    __builtin_amdgcn_s_wait_tensorcnt(0);
  }
#elif USE_ASYNC
  // ---- async global->LDS: 256 threads x 32 B = 8 KB -----------------------
  {
    const char* gp = (const char*)w1frag + tid * 32;
    char* sp = (char*)w1s + tid * 32;
    __builtin_amdgcn_global_load_async_to_lds_b128(
        (__attribute__((address_space(1))) void*)(void*)gp,
        (__attribute__((address_space(3))) void*)(void*)sp, 0, 0);
    __builtin_amdgcn_global_load_async_to_lds_b128(
        (__attribute__((address_space(1))) void*)(void*)(gp + 16),
        (__attribute__((address_space(3))) void*)(void*)(sp + 16), 0, 0);
    __builtin_amdgcn_s_wait_asynccnt(0);
  }
#else
  {
    const unsigned int* src = (const unsigned int*)w1frag;
    unsigned int* dst = (unsigned int*)w1s;
#pragma unroll
    for (int j = 0; j < 8; ++j) dst[tid + 256 * j] = src[tid + 256 * j];
  }
#endif
  __syncthreads();

  const int wave = tid >> 5, lane = tid & 31;
  const int sel = lane >> 4, nlo = lane & 15;
  const size_t rowBase = ((size_t)blockIdx.x * 8 + wave) * 16;

  // A fragments: 16x128 bf16 tile, 4 K-steps of 32.
  // lane<16: row=nlo, K {0..7}/{16..23}+32k ; lane>=16: row=nlo, K {8..15}/{24..31}+32k
  const char* rowp = (const char*)feats + (rowBase + nlo) * (F2PAD * 2) + sel * 16;
  __builtin_prefetch(rowp + 8 * 16 * (F2PAD * 2), 0, 1);  // next block's rows
  v16bf a[4];
#pragma unroll
  for (int k = 0; k < 4; ++k) {
    BFrag t;
    t.q[0] = *(const uint4*)(rowp + k * 64);
    t.q[1] = *(const uint4*)(rowp + k * 64 + 32);
    a[k] = t.v;
  }

  v8f acc0 = {0.f, 0.f, 0.f, 0.f, 0.f, 0.f, 0.f, 0.f};
  v8f acc1 = {0.f, 0.f, 0.f, 0.f, 0.f, 0.f, 0.f, 0.f};
#pragma unroll
  for (int k = 0; k < 4; ++k) {
    BFrag bA, bB;
    const uint4* p0 = (const uint4*)&w1s[(k * 2 + 0) * 512 + lane * 16];
    const uint4* p1 = (const uint4*)&w1s[(k * 2 + 1) * 512 + lane * 16];
    bA.q[0] = p0[0]; bA.q[1] = p0[1];
    bB.q[0] = p1[0]; bB.q[1] = p1[1];
    acc0 = __builtin_amdgcn_wmma_f32_16x16x32_bf16(false, a[k], false, bA.v,
                                                   (short)0, acc0, false, false);
    acc1 = __builtin_amdgcn_wmma_f32_16x16x32_bf16(false, a[k], false, bB.v,
                                                   (short)0, acc1, false, false);
  }

  // Epilogue: bias + SELU + layer-2 partial dot (each lane owns column n).
  float b1n0 = b1[nlo],      b1n1 = b1[16 + nlo];
  float w2n0 = W2[nlo],      w2n1 = W2[16 + nlo];
  float part[8];
#pragma unroll
  for (int r = 0; r < 8; ++r) {
    float h0 = seluf(acc0[r] + b1n0);
    float h1 = seluf(acc1[r] + b1n1);
    part[r] = h0 * w2n0 + h1 * w2n1;
  }
  // reduce across the 16 lanes of each half-wave (bit4 untouched)
#pragma unroll
  for (int off = 1; off < 16; off <<= 1) {
#pragma unroll
    for (int r = 0; r < 8; ++r) part[r] += __shfl_xor(part[r], off, 32);
  }

  if (nlo == 0) {
    // lane 0 -> rows rowBase..+7 ; lane 16 -> rows rowBase+8..+15 (C layout)
    float bb = b2[0];
    float pg = 0.f;
    size_t m0 = rowBase + (size_t)sel * 8;
#pragma unroll
    for (int r = 0; r < 8; ++r) {
      size_t i = m0 + r;
      float p = sigmoidf_(part[r] + bb);
      probs[i] = p;
      int s = (ud[i] < p) ? 1 : 0;
      splits[i] = s;
      pg += mask[i] * (s ? p : 1.f - p);
    }
    atomicAdd(pgAcc, pg);
  }
}

// ---------------------------------------------------------------------------
// Child segment (post-split) sums & counts -> leaf detection + leaf means.
// ---------------------------------------------------------------------------
__global__ void child_accum(const float* __restrict__ Xn, const int* __restrict__ seg,
                            const int* __restrict__ splits,
                            float* __restrict__ childSum, int* __restrict__ childCnt) {
  int i = blockIdx.x * 256 + threadIdx.x;
  if (i >= NPTS) return;
  int s = seg[i];
  if (s < 0) return;
  int c = 2 * s + splits[i];
  atomicAdd(&childCnt[c], 1);
  const float* x = &Xn[(size_t)i * FPAD];
  float* o = &childSum[(size_t)c * FPAD];
  for (int f = 0; f < FDIM; ++f) atomicAdd(&o[f], x[f]);
}

// Sequential scan over children in (l_split, r_split) order -> deterministic
// compact subtree ids and leaf/node ids matching the reference replay order.
__global__ void classify_scan(const int* __restrict__ childCnt, int* __restrict__ newSegId,
                              int* __restrict__ leafId, int* __restrict__ leafNode,
                              int* __restrict__ ctrlI) {
  if (threadIdx.x || blockIdx.x) return;
  int K  = ctrlI[CI_NUMSEG];
  int K2 = 2 * K;
  if (K2 > CMAX) K2 = CMAX;
  ctrlI[CI_K2] = K2;
  int nodeBase = ctrlI[CI_NODE];
  int nSub = 0, nLeaf = 0;
  for (int c = 0; c < K2; ++c) {
    int cnt = childCnt[c];
    if (cnt <= 0) { newSegId[c] = -1; leafId[c] = -1; continue; }
    if (cnt < MAXN) {
      leafId[c] = nLeaf; leafNode[c] = nodeBase + nLeaf; newSegId[c] = -1; ++nLeaf;
    } else {
      newSegId[c] = nSub++; leafId[c] = -1;
    }
  }
  ctrlI[CI_NODE]    = nodeBase + nLeaf;
  ctrlI[CI_NUMSEG]  = nSub;
  ctrlI[CI_NUMLEAF] = nLeaf;
}

__global__ void emit_leaf_nodes(const float* __restrict__ childSum, const int* __restrict__ childCnt,
                                const int* __restrict__ leafId, const int* __restrict__ leafNode,
                                const int* __restrict__ ctrlI, float* __restrict__ dout) {
  size_t t = (size_t)blockIdx.x * blockDim.x + threadIdx.x;
  int f = (int)(t & 63);
  int c = (int)(t >> 6);
  if (c >= ctrlI[CI_K2]) return;
  if (leafId[c] < 0) return;
  int row = leafNode[c];
  int cc  = childCnt[c] > 0 ? childCnt[c] : 1;
  float v = (f < FDIM) ? childSum[(size_t)c * FPAD + f] / (float)cc : 1.f;
  dout[(size_t)row * FPAD + f] = v;
}

// Relabel points, drop leaf points from the active set, emit tree edges and
// leaf membership lists.
__global__ void relabel(int* __restrict__ seg, const int* __restrict__ splits,
                        float* __restrict__ mask,
                        const int* __restrict__ newSegId, const int* __restrict__ leafId,
                        const int* __restrict__ leafNode, const int* __restrict__ levelNode,
                        int* __restrict__ leafFill, int* __restrict__ leafMembers,
                        int* __restrict__ ctrlI, int* __restrict__ edgeBuf) {
  int i = blockIdx.x * 256 + threadIdx.x;
  if (i >= NPTS) return;
  int s = seg[i];
  if (s < 0) return;
  // edges point <-> this level's subtree node
  int ln = levelNode[s];
  unsigned e = atomicAdd((unsigned*)&ctrlI[CI_EDGE], 2u);
  if (e + 1 < (unsigned)EMAX) {
    edgeBuf[2 * e + 0] = i;  edgeBuf[2 * e + 1] = ln;
    edgeBuf[2 * e + 2] = ln; edgeBuf[2 * e + 3] = i;
  }
  int c = 2 * s + splits[i];
  int lid = leafId[c];
  if (lid >= 0) {
    mask[i] = 0.f;
    seg[i]  = -1;
    int lnode = leafNode[c];
    unsigned e2 = atomicAdd((unsigned*)&ctrlI[CI_EDGE], 2u);
    if (e2 + 1 < (unsigned)EMAX) {
      edgeBuf[2 * e2 + 0] = i;     edgeBuf[2 * e2 + 1] = lnode;
      edgeBuf[2 * e2 + 2] = lnode; edgeBuf[2 * e2 + 3] = i;
    }
    int slot = atomicAdd(&leafFill[lid], 1);
    if (slot < MAXN) leafMembers[lid * MAXN + slot] = i;
  } else {
    seg[i] = newSegId[c];
  }
}

// All-pairs (dense) edges inside each new leaf.
__global__ void leaf_pairs(const int* __restrict__ leafFill, const int* __restrict__ leafMembers,
                           int* __restrict__ ctrlI, int* __restrict__ edgeBuf) {
  int j = blockIdx.x;
  if (j >= ctrlI[CI_NUMLEAF]) return;
  int nn = leafFill[j];
  if (nn <= 0) return;
  if (nn > MAXN) nn = MAXN;
  int total = nn * nn;
  __shared__ unsigned base;
  if (threadIdx.x == 0) base = atomicAdd((unsigned*)&ctrlI[CI_EDGE], (unsigned)total);
  __syncthreads();
  for (int t = threadIdx.x; t < total; t += blockDim.x) {
    unsigned e = base + (unsigned)t;
    if (e >= (unsigned)EMAX) break;
    edgeBuf[2 * e + 0] = leafMembers[j * MAXN + t / nn];
    edgeBuf[2 * e + 1] = leafMembers[j * MAXN + t % nn];
  }
}

// Place COO indices / values / pg_sum / edge count after the node block.
__global__ void finalize(const int* __restrict__ ctrlI, const float* __restrict__ ctrlF,
                         const int* __restrict__ edgeBuf, float* __restrict__ dout,
                         long long out_size) {
  long long E = ctrlI[CI_EDGE];
  if (E > (long long)EMAX) E = EMAX;
  long long R = ctrlI[CI_NODE];
  long long t = (long long)blockIdx.x * blockDim.x + threadIdx.x;
  long long* idx = (long long*)(dout + R * FPAD);
  float* vv = (float*)(idx + 2 * E);
  if (t < E) {
    idx[t]     = (long long)edgeBuf[2 * t + 0];
    idx[E + t] = (long long)edgeBuf[2 * t + 1];
    vv[t] = 1.0f;
  }
  if (t == 0) {
    vv[E]     = ctrlF[0];
    vv[E + 1] = (float)E;
    if (out_size >= 2) {
      dout[out_size - 2] = ctrlF[0];
      dout[out_size - 1] = (float)E;
    }
  }
}

// ---------------------------------------------------------------------------
extern "C" void kernel_launch(void* const* d_in, const int* in_sizes, int n_in,
                              void* d_out, int out_size, void* d_ws, size_t ws_size,
                              hipStream_t stream) {
  (void)in_sizes; (void)n_in; (void)ws_size;
  const float* X_in = (const float*)d_in[0];
  const float* u    = (const float*)d_in[1];
  const float* g    = (const float*)d_in[2];
  const float* bta  = (const float*)d_in[3];
  const float* W1   = (const float*)d_in[4];
  const float* b1   = (const float*)d_in[5];
  const float* W2   = (const float*)d_in[6];
  const float* b2   = (const float*)d_in[7];
  float* dout = (float*)d_out;

  char* w = (char*)d_ws;
  size_t off = 0;
  auto alloc = [&](size_t bytes) -> char* {
    char* p = w + off;
    off = (off + bytes + 255) & ~(size_t)255;
    return p;
  };

  float* bnSum  = (float*)alloc(64 * 4);
  float* bnSq   = (float*)alloc(64 * 4);
  float* ctrlF  = (float*)alloc(16 * 4);
  int*   ctrlI  = (int*)alloc(16 * 4);
  float* Xn     = (float*)alloc((size_t)NPTS * FPAD * 4);
  unsigned short* feats = (unsigned short*)alloc((size_t)NPTS * F2PAD * 2);
  int*   seg    = (int*)alloc((size_t)NPTS * 4);
  int*   splits = (int*)alloc((size_t)NPTS * 4);
  float* mask   = (float*)alloc((size_t)NPTS * 4);
  float* probs  = (float*)alloc((size_t)NPTS * 4);
  float* segSum = (float*)alloc((size_t)KMAX * FPAD * 4);
  int*   segCnt = (int*)alloc((size_t)KMAX * 4);
  int*   levelNode = (int*)alloc((size_t)KMAX * 4);
  float* childSum  = (float*)alloc((size_t)CMAX * FPAD * 4);
  int*   childCnt  = (int*)alloc((size_t)CMAX * 4);
  int*   newSegId  = (int*)alloc((size_t)CMAX * 4);
  int*   leafId    = (int*)alloc((size_t)CMAX * 4);
  int*   leafNode  = (int*)alloc((size_t)CMAX * 4);
  int*   leafFill  = (int*)alloc((size_t)CMAX * 4);
  int*   leafMembers = (int*)alloc((size_t)CMAX * MAXN * 4);
  unsigned short* w1frag = (unsigned short*)alloc(4096 * 2);
  int*   edgeBuf = (int*)alloc((size_t)EMAX * 2 * 4);

  hipMemsetAsync(bnSum, 0, 64 * 4, stream);
  hipMemsetAsync(bnSq,  0, 64 * 4, stream);
  init_state<<<(NPTS + 255) / 256, 256, 0, stream>>>(seg, mask, ctrlI, ctrlF);
  bn_reduce<<<FDIM, 256, 0, stream>>>(X_in, bnSum, bnSq);
  bn_normalize<<<(NPTS * FPAD) / 256, 256, 0, stream>>>(X_in, bnSum, bnSq, g, bta,
                                                        Xn, feats, dout);
  prep_w1frag<<<16, 256, 0, stream>>>(W1, w1frag);

  for (int d = 0; d < NLEV; ++d) {
    hipMemsetAsync(segSum,   0, (size_t)KMAX * FPAD * 4, stream);
    hipMemsetAsync(segCnt,   0, (size_t)KMAX * 4, stream);
    hipMemsetAsync(childSum, 0, (size_t)CMAX * FPAD * 4, stream);
    hipMemsetAsync(childCnt, 0, (size_t)CMAX * 4, stream);
    hipMemsetAsync(leafFill, 0, (size_t)CMAX * 4, stream);

    seg_accum<<<NPTS / 256, 256, 0, stream>>>(Xn, seg, segSum, segCnt);
    level_node_base<<<1, 32, 0, stream>>>(ctrlI);
    emit_level_nodes<<<(KMAX * FPAD) / 256, 256, 0, stream>>>(segSum, segCnt, ctrlI,
                                                              levelNode, dout);
    agg_write<<<(NPTS * FPAD) / 256, 256, 0, stream>>>(segSum, segCnt, seg, feats);
    mlp_wmma<<<NPTS / 128, 256, 0, stream>>>(feats, w1frag, b1, W2, b2,
                                             u + (size_t)d * NPTS, mask,
                                             probs, splits, ctrlF);
    child_accum<<<NPTS / 256, 256, 0, stream>>>(Xn, seg, splits, childSum, childCnt);
    classify_scan<<<1, 32, 0, stream>>>(childCnt, newSegId, leafId, leafNode, ctrlI);
    emit_leaf_nodes<<<(CMAX * FPAD) / 256, 256, 0, stream>>>(childSum, childCnt, leafId,
                                                             leafNode, ctrlI, dout);
    relabel<<<NPTS / 256, 256, 0, stream>>>(seg, splits, mask, newSegId, leafId, leafNode,
                                            levelNode, leafFill, leafMembers, ctrlI, edgeBuf);
    leaf_pairs<<<KMAX, 128, 0, stream>>>(leafFill, leafMembers, ctrlI, edgeBuf);
  }

  finalize<<<EMAX / 256, 256, 0, stream>>>(ctrlI, ctrlF, edgeBuf, dout, (long long)out_size);
}
